// SleepOptomizer1_12850542150188
// MI455X (gfx1250) — compile-verified, matrix operand order checked by reading, not run on hardware
//
#include <hip/hip_runtime.h>

// ---------------------------------------------------------------------------
// Types for CDNA5 WMMA (wave32): bf16 A/B fragments (8 VGPRs), f32 C/D (8 VGPRs)
// ---------------------------------------------------------------------------
typedef __attribute__((ext_vector_type(16))) __bf16 bf16x16;
typedef __attribute__((ext_vector_type(8)))  __bf16 bf16x8;
typedef __attribute__((ext_vector_type(8)))  float  f32x8;
typedef __attribute__((ext_vector_type(4)))  int    v4i;

#define WMMA_BF16(a, b, c) \
  __builtin_amdgcn_wmma_f32_16x16x32_bf16(false, (a), false, (b), (short)0, (c), false, false)

// CDNA5 async global->LDS path (ASYNCcnt-tracked), guarded so the file compiles
// on toolchains that lack the builtins.
#if defined(__has_builtin)
#if __has_builtin(__builtin_amdgcn_global_load_async_to_lds_b128) && \
    __has_builtin(__builtin_amdgcn_s_wait_asynccnt)
#define USE_ASYNC_LDS 1
#endif
#endif
#ifndef USE_ASYNC_LDS
#define USE_ASYNC_LDS 0
#endif

// Fragment loaders. ISA 16-bit A layout (16x32 MxK, wave32):
//   lane = 16*khalf + m ; VGPR0..3 hold K = khalf*8 + {0..7}, VGPR4..7 hold
//   K = 16 + khalf*8 + {0..7}  -> two contiguous 8-element runs = two b128 loads.
// B (32x16 KxN) is the symmetric layout with n in place of m, sourced from a
// pre-transposed [N][K] buffer so its runs are contiguous as well.
__device__ __forceinline__ bf16x16 frag_bf16(const __bf16* base, int ld, int row, int k0, int kh) {
  const __bf16* p = base + (size_t)row * ld + k0 + kh * 8;
  bf16x8 lo = *(const bf16x8*)p;
  bf16x8 hi = *(const bf16x8*)(p + 16);
  return __builtin_shufflevector(lo, hi, 0, 1, 2, 3, 4, 5, 6, 7, 8, 9, 10, 11, 12, 13, 14, 15);
}

__device__ __forceinline__ bf16x16 frag_lds(const __bf16* p) {
  bf16x8 lo = *(const bf16x8*)p;
  bf16x8 hi = *(const bf16x8*)(p + 16);
  return __builtin_shufflevector(lo, hi, 0, 1, 2, 3, 4, 5, 6, 7, 8, 9, 10, 11, 12, 13, 14, 15);
}

__device__ __forceinline__ bf16x16 frag_from_f32(const float* base, int ld, int row, int k0, int kh) {
  const float* p = base + (size_t)row * ld + k0 + kh * 8;
  float4 f0 = *(const float4*)(p);
  float4 f1 = *(const float4*)(p + 4);
  float4 f2 = *(const float4*)(p + 16);
  float4 f3 = *(const float4*)(p + 20);
  bf16x16 a;
  a[0] = (__bf16)f0.x;  a[1] = (__bf16)f0.y;  a[2] = (__bf16)f0.z;  a[3] = (__bf16)f0.w;
  a[4] = (__bf16)f1.x;  a[5] = (__bf16)f1.y;  a[6] = (__bf16)f1.z;  a[7] = (__bf16)f1.w;
  a[8] = (__bf16)f2.x;  a[9] = (__bf16)f2.y;  a[10] = (__bf16)f2.z; a[11] = (__bf16)f2.w;
  a[12] = (__bf16)f3.x; a[13] = (__bf16)f3.y; a[14] = (__bf16)f3.z; a[15] = (__bf16)f3.w;
  return a;
}

// ---------------------------------------------------------------------------
// Weight transpose+convert: src f32 [K][N] -> dst bf16 [Npad][K], zero pad.
// ---------------------------------------------------------------------------
__global__ void k_transpose_bf16(const float* __restrict__ src, __bf16* __restrict__ dst,
                                 int K, int N, int Npad) {
  int idx = blockIdx.x * 256 + threadIdx.x;
  if (idx >= Npad * K) return;
  int n = idx / K, k = idx - n * K;
  float v = (n < N) ? src[(size_t)k * N + n] : 0.0f;
  dst[idx] = (__bf16)v;
}

// ---------------------------------------------------------------------------
// GEMM 1: xw = inputs[65536,512] @ Wx[512,1024] + b_rnn  -> bf16 [65536,1024]
// One wave computes a 16x64 tile (4 accumulators). Streams inputs once,
// converting f32->bf16 inline (cheaper than a separate convert pass).
// ---------------------------------------------------------------------------
__global__ void __launch_bounds__(256) k_xw_gemm(const float* __restrict__ X,
                                                 const __bf16* __restrict__ WxT,
                                                 const float* __restrict__ brnn,
                                                 __bf16* __restrict__ xw) {
  int wave = (blockIdx.x * blockDim.x + threadIdx.x) >> 5;
  int lane = threadIdx.x & 31;
  int mt = wave >> 4, nt = wave & 15;           // 4096 x 16 tiles
  int M0 = mt * 16, N0 = nt * 64;
  int ml = lane & 15, kh = lane >> 4;
  f32x8 acc[4] = {};
  for (int k0 = 0; k0 < 512; k0 += 32) {
    bf16x16 a = frag_from_f32(X, 512, M0 + ml, k0, kh);
#pragma unroll
    for (int j = 0; j < 4; ++j) {
      bf16x16 b = frag_bf16(WxT, 512, N0 + j * 16 + ml, k0, kh);
      acc[j] = WMMA_BF16(a, b, acc[j]);
    }
  }
  int g = lane >> 4, n = lane & 15;
#pragma unroll
  for (int j = 0; j < 4; ++j)
#pragma unroll
    for (int r = 0; r < 8; ++r) {
      int row = M0 + r + 8 * g;
      int col = N0 + j * 16 + n;
      xw[(size_t)row * 1024 + col] = (__bf16)(acc[j][r] + brnn[col]);
    }
}

// ---------------------------------------------------------------------------
// RNN step t: hnew = relu(xw[:,t,:] + hprev @ Wh)   (M=256, N=K=1024)
// Block = 8 waves = 8 M-tiles sharing one 64-wide N strip of Wh, staged into
// LDS in double-buffered 16KB K-chunks via the CDNA5 async global->LDS DMA.
// ---------------------------------------------------------------------------
__global__ void __launch_bounds__(256) k_rnn_step(const __bf16* __restrict__ hprev,
                                                  const __bf16* __restrict__ WhT,
                                                  const __bf16* __restrict__ xw, int t,
                                                  __bf16* __restrict__ hnew) {
  __shared__ __bf16 WhS[2][64][128];   // 2 x 16KB
  int tid = threadIdx.x;
  int wv = tid >> 5, lane = tid & 31;
  int nt = blockIdx.x & 15;            // N64 strip
  int mh = blockIdx.x >> 4;            // which half of the 16 M-tiles
  int N0 = nt * 64;
  int M0 = (mh * 8 + wv) * 16;
  int ml = lane & 15, kh = lane >> 4;

  // Stage one 64x128 bf16 chunk of WhT into LDS: 1024 x 16B vectors, 4/thread.
  auto stage = [&](int chunk, int buf) {
    int kbase = chunk * 128;
#pragma unroll
    for (int i = 0; i < 4; ++i) {
      int v = tid + i * 256;
      int row = v >> 4;                // 16 vectors per 128-elem row
      int col = (v & 15) * 8;
      const __bf16* g = WhT + (size_t)(N0 + row) * 1024 + kbase + col;
      __bf16* l = &WhS[buf][row][col];
#if USE_ASYNC_LDS
      __builtin_amdgcn_global_load_async_to_lds_b128(
          (__attribute__((address_space(1))) v4i*)(v4i*)const_cast<__bf16*>(g),
          (__attribute__((address_space(3))) v4i*)(v4i*)l,
          0, 0);
#else
      *(bf16x8*)l = *(const bf16x8*)g;
#endif
    }
  };

  stage(0, 0);
#if USE_ASYNC_LDS
  __builtin_amdgcn_s_wait_asynccnt(0);
#endif
  __syncthreads();

  f32x8 acc[4] = {};
  for (int c = 0; c < 8; ++c) {
    if (c + 1 < 8) stage(c + 1, (c + 1) & 1);   // prefetch next chunk
    int buf = c & 1;
    int kglob = c * 128;
#pragma unroll
    for (int kk = 0; kk < 128; kk += 32) {
      bf16x16 a = frag_bf16(hprev, 1024, M0 + ml, kglob + kk, kh);
#pragma unroll
      for (int j = 0; j < 4; ++j) {
        bf16x16 b = frag_lds(&WhS[buf][j * 16 + ml][kk + kh * 8]);
        acc[j] = WMMA_BF16(a, b, acc[j]);
      }
    }
#if USE_ASYNC_LDS
    __builtin_amdgcn_s_wait_asynccnt(0);
#endif
    __syncthreads();
  }

  int g = lane >> 4, n = lane & 15;
#pragma unroll
  for (int j = 0; j < 4; ++j)
#pragma unroll
    for (int r = 0; r < 8; ++r) {
      int row = M0 + r + 8 * g;                  // batch index b
      int col = N0 + j * 16 + n;
      float v = acc[j][r] + (float)xw[((size_t)row * 256 + t) * 1024 + col];
      hnew[(size_t)row * 1024 + col] = (__bf16)fmaxf(v, 0.0f);
    }
}

// ---------------------------------------------------------------------------
// Dense: out = (relu?)(A[256,1024] @ B[1024,1024] + bias). Writes bf16 and/or f32.
// ---------------------------------------------------------------------------
__global__ void __launch_bounds__(256) k_dense(const __bf16* __restrict__ A,
                                               const __bf16* __restrict__ BT,
                                               const float* __restrict__ bias, int nbias,
                                               int do_relu, __bf16* __restrict__ out_bf,
                                               float* __restrict__ out_f) {
  int wave = (blockIdx.x * blockDim.x + threadIdx.x) >> 5;
  int lane = threadIdx.x & 31;
  int mt = wave >> 4, nt = wave & 15;
  int M0 = mt * 16, N0 = nt * 64;
  int ml = lane & 15, kh = lane >> 4;
  f32x8 acc[4] = {};
  for (int k0 = 0; k0 < 1024; k0 += 32) {
    bf16x16 a = frag_bf16(A, 1024, M0 + ml, k0, kh);
#pragma unroll
    for (int j = 0; j < 4; ++j) {
      bf16x16 b = frag_bf16(BT, 1024, N0 + j * 16 + ml, k0, kh);
      acc[j] = WMMA_BF16(a, b, acc[j]);
    }
  }
  int g = lane >> 4, n = lane & 15;
#pragma unroll
  for (int j = 0; j < 4; ++j)
#pragma unroll
    for (int r = 0; r < 8; ++r) {
      int row = M0 + r + 8 * g;
      int col = N0 + j * 16 + n;
      float v = acc[j][r] + ((col < nbias) ? bias[col] : 0.0f);
      if (do_relu) v = fmaxf(v, 0.0f);
      if (out_bf) out_bf[(size_t)row * 1024 + col] = (__bf16)v;
      if (out_f)  out_f[(size_t)row * 1024 + col] = v;
    }
}

// ---------------------------------------------------------------------------
// JAX threefry2x32 (key = (0,42)) for faithful categorical sampling.
// ---------------------------------------------------------------------------
__device__ __forceinline__ void threefry2x32(unsigned k0, unsigned k1, unsigned x0, unsigned x1,
                                             unsigned& o0, unsigned& o1) {
  unsigned ks[3] = {k0, k1, k0 ^ k1 ^ 0x1BD11BDAu};
  const int R[8] = {13, 15, 26, 6, 17, 29, 16, 24};
  x0 += ks[0]; x1 += ks[1];
#pragma unroll
  for (int g = 0; g < 5; ++g) {
    const int* r = (g & 1) ? (R + 4) : R;
#pragma unroll
    for (int i = 0; i < 4; ++i) {
      x0 += x1;
      x1 = (x1 << r[i]) | (x1 >> (32 - r[i]));
      x1 ^= x0;
    }
    x0 += ks[(g + 1) % 3];
    x1 += ks[(g + 2) % 3] + (unsigned)(g + 1);
  }
  o0 = x0; o1 = x1;
}

__device__ __forceinline__ float gumbel_for(unsigned j) {
  const unsigned half = 128000u;   // B*A/2 = 256*1000/2
  unsigned o0, o1, bits;
  if (j < half) { threefry2x32(0u, 42u, j, j + half, o0, o1); bits = o0; }
  else          { threefry2x32(0u, 42u, j - half, j, o0, o1); bits = o1; }
  float f = __uint_as_float((bits >> 9) | 0x3f800000u) - 1.0f;   // [0,1)
  const float tiny = 1.1754943508222875e-38f;
  float u = fmaxf(tiny, f * (1.0f - tiny) + tiny);
  return -logf(-logf(u));
}

// ---------------------------------------------------------------------------
// Per-row: softmax over 1000 logits, gumbel-argmax on the *probabilities*
// (reference passes probs to categorical), and per-row CE loss term.
// ---------------------------------------------------------------------------
__global__ void k_finalize(const float* __restrict__ logits, int* __restrict__ action,
                           float* __restrict__ rowloss) {
  __shared__ float sf[256];
  __shared__ int   si[256];
  int b = blockIdx.x, tid = threadIdx.x;
  const float* row = logits + (size_t)b * 1024;

  // max
  float lmax = -3.402823466e+38f;
  for (int a = tid; a < 1000; a += 256) lmax = fmaxf(lmax, row[a]);
  sf[tid] = lmax; __syncthreads();
  for (int s = 128; s > 0; s >>= 1) { if (tid < s) sf[tid] = fmaxf(sf[tid], sf[tid + s]); __syncthreads(); }
  float mx = sf[0]; __syncthreads();

  // sum of exp
  float lsum = 0.0f;
  for (int a = tid; a < 1000; a += 256) lsum += expf(row[a] - mx);
  sf[tid] = lsum; __syncthreads();
  for (int s = 128; s > 0; s >>= 1) { if (tid < s) sf[tid] += sf[tid + s]; __syncthreads(); }
  float inv = 1.0f / sf[0]; __syncthreads();

  // gumbel-argmax over p + g (first index wins ties)
  float bestv = -3.402823466e+38f; int besti = 0x7fffffff;
  for (int a = tid; a < 1000; a += 256) {
    float p = expf(row[a] - mx) * inv;
    float s = p + gumbel_for((unsigned)(b * 1000 + a));
    if (s > bestv) { bestv = s; besti = a; }
  }
  sf[tid] = bestv; si[tid] = besti; __syncthreads();
  for (int s = 128; s > 0; s >>= 1) {
    if (tid < s) {
      float v2 = sf[tid + s]; int i2 = si[tid + s];
      if (v2 > sf[tid] || (v2 == sf[tid] && i2 < si[tid])) { sf[tid] = v2; si[tid] = i2; }
    }
    __syncthreads();
  }
  if (tid == 0) {
    int a = si[0];
    action[b] = a;
    float p = expf(row[a] - mx) * inv;
    p = fminf(fmaxf(p, 1e-7f), 1.0f - 1e-7f);
    rowloss[b] = -logf(p);
  }
}

__global__ void k_loss_reduce(const float* __restrict__ rowloss, float* __restrict__ out) {
  __shared__ float sf[256];
  int tid = threadIdx.x;
  sf[tid] = rowloss[tid]; __syncthreads();
  for (int s = 128; s > 0; s >>= 1) { if (tid < s) sf[tid] += sf[tid + s]; __syncthreads(); }
  if (tid == 0) out[0] = sf[0] / 256.0f;
}

// ---------------------------------------------------------------------------
// Host launcher
// ---------------------------------------------------------------------------
extern "C" void kernel_launch(void* const* d_in, const int* in_sizes, int n_in,
                              void* d_out, int out_size, void* d_ws, size_t ws_size,
                              hipStream_t stream) {
  const float* inputs = (const float*)d_in[0];   // [256,256,512]
  const float* Wx     = (const float*)d_in[1];   // [512,1024]
  const float* Wh     = (const float*)d_in[2];   // [1024,1024]
  const float* b_rnn  = (const float*)d_in[3];   // [1024]
  const float* W1     = (const float*)d_in[4];   // [1024,1024]
  const float* b1     = (const float*)d_in[5];   // [1024]
  const float* W2     = (const float*)d_in[6];   // [1024,1000]
  const float* b2     = (const float*)d_in[7];   // [1000]

  char* w = (char*)d_ws;
  size_t oWx  = 0;                              // WxT bf16 [1024][512]
  size_t oWh  = oWx  + (size_t)1024 * 512 * 2;  // WhT bf16 [1024][1024]
  size_t oW1  = oWh  + (size_t)1024 * 1024 * 2;
  size_t oW2  = oW1  + (size_t)1024 * 1024 * 2; // W2T bf16 [1024pad][1024]
  size_t oXW  = oW2  + (size_t)1024 * 1024 * 2; // xw bf16 [65536][1024]
  size_t oHA  = oXW  + (size_t)65536 * 1024 * 2;
  size_t oHB  = oHA  + (size_t)256 * 1024 * 2;
  size_t oHID = oHB  + (size_t)256 * 1024 * 2;
  size_t oLG  = oHID + (size_t)256 * 1024 * 2;  // logits f32 [256][1024]
  size_t oRL  = oLG  + (size_t)256 * 1024 * 4;  // rowloss f32 [256]
  size_t need = oRL + 256 * 4;
  if (ws_size < need) return;

  __bf16* WxT  = (__bf16*)(w + oWx);
  __bf16* WhT  = (__bf16*)(w + oWh);
  __bf16* W1T  = (__bf16*)(w + oW1);
  __bf16* W2T  = (__bf16*)(w + oW2);
  __bf16* xw   = (__bf16*)(w + oXW);
  __bf16* hA   = (__bf16*)(w + oHA);
  __bf16* hB   = (__bf16*)(w + oHB);
  __bf16* hid  = (__bf16*)(w + oHID);
  float*  lgts = (float*)(w + oLG);
  float*  rlos = (float*)(w + oRL);

  // 1) Weight transposes to bf16 [N][K]
  k_transpose_bf16<<<(1024 * 512 + 255) / 256, 256, 0, stream>>>(Wx, WxT, 512, 1024, 1024);
  k_transpose_bf16<<<(1024 * 1024 + 255) / 256, 256, 0, stream>>>(Wh, WhT, 1024, 1024, 1024);
  k_transpose_bf16<<<(1024 * 1024 + 255) / 256, 256, 0, stream>>>(W1, W1T, 1024, 1024, 1024);
  k_transpose_bf16<<<(1024 * 1024 + 255) / 256, 256, 0, stream>>>(W2, W2T, 1024, 1000, 1024);

  // 2) xw = inputs @ Wx + b_rnn   (65536 waves of 16x64 tiles, 8 waves/block)
  k_xw_gemm<<<8192, 256, 0, stream>>>(inputs, WxT, b_rnn, xw);

  // 3) recurrence, h0 = 0
  (void)hipMemsetAsync(hA, 0, (size_t)256 * 1024 * 2, stream);
  __bf16* hp = hA;
  __bf16* hn = hB;
  for (int t = 0; t < 256; ++t) {
    k_rnn_step<<<32, 256, 0, stream>>>(hp, WhT, xw, t, hn);
    __bf16* tmp = hp; hp = hn; hn = tmp;   // hp now holds h_t
  }

  // 4) hid = relu(h_last @ W1 + b1)
  k_dense<<<32, 256, 0, stream>>>(hp, W1T, b1, 1024, 1, hid, nullptr);
  // 5) logits = hid @ W2 + b2 (N padded to 1024, pad weight rows are zero)
  k_dense<<<32, 256, 0, stream>>>(hid, W2T, b2, 1000, 0, nullptr, lgts);

  // 6) softmax + threefry gumbel-argmax + CE loss
  k_finalize<<<256, 256, 0, stream>>>(lgts, (int*)d_out, rlos);
  k_loss_reduce<<<1, 256, 0, stream>>>(rlos, (float*)d_out + 256);
}
